// SimpleGate_83674552861192
// MI455X (gfx1250) — matrix operand order repaired and checked by reading, not compiled, not run on hardware
//
#include <hip/hip_runtime.h>

// Problem constants (match the reference)
#define TOKENS   16384
#define KDIM     4096
#define HID      256
#define NEXP     64

// Tiling
#define BLOCK_M  64                 // tokens per workgroup
#define KSTEP    64                 // staged K-chunk (2 WMMA K-substeps)
#define NSTEP    (KDIM / KSTEP)     // 64 staged steps for GEMM1
#define NKC2     (HID / 32)         // 8 K-steps for GEMM2

// Padded LDS row strides (bank-conflict avoidance on 64 banks x 4B)
#define AP  72    // lds_a row stride in shorts (36 dwords, gcd(36,64)=4)
#define HP  264   // lds_h row stride in shorts (132 dwords)
#define LP  65    // lds_logits row stride in floats

typedef __attribute__((ext_vector_type(16))) __bf16 v16bf;
typedef __attribute__((ext_vector_type(2)))  __bf16 v2bf;
typedef __attribute__((ext_vector_type(8)))  float  v8f;

union Frag32B {
    uint4 u[2];
    v16bf v;
};

// hardware f32->bf16 (clang lowers fptrunc to v_cvt_pk_bf16_f32-class ops)
__device__ __forceinline__ unsigned int pkbf(float lo, float hi) {
    v2bf t;
    t[0] = (__bf16)lo;
    t[1] = (__bf16)hi;
    return __builtin_bit_cast(unsigned int, t);
}
__device__ __forceinline__ unsigned short bf1(float f) {
    __bf16 t = (__bf16)f;
    return __builtin_bit_cast(unsigned short, t);
}

// ---- one-off weight conversion kernels: f32 -> bf16, transposed (K contiguous) ----
__global__ void cvt_w1_kernel(const float* __restrict__ w1, unsigned short* __restrict__ w1t) {
    int idx = blockIdx.x * blockDim.x + threadIdx.x;   // over KDIM*HID
    int k = idx >> 8;          // HID == 256
    int n = idx & 255;
    w1t[(size_t)n * KDIM + k] = bf1(w1[idx]);
}

__global__ void cvt_w2_kernel(const float* __restrict__ w2, unsigned short* __restrict__ w2t) {
    int idx = blockIdx.x * blockDim.x + threadIdx.x;   // over HID*NEXP
    int k = idx >> 6;          // NEXP == 64
    int e = idx & 63;
    w2t[(size_t)e * HID + k] = bf1(w2[idx]);
}

// ---- fused router-gate kernel ----
__global__ void __launch_bounds__(256)
router_gate_kernel(const float* __restrict__ x,
                   const float* __restrict__ b1,
                   const float* __restrict__ b2,
                   const unsigned short* __restrict__ w1t,   // [HID][KDIM] bf16
                   const unsigned short* __restrict__ w2t,   // [NEXP][HID] bf16
                   float* __restrict__ out) {
    __shared__ __align__(16) unsigned short lds_a[2][BLOCK_M * AP];   // 2 x 9 KB
    __shared__ __align__(16) unsigned short lds_h[BLOCK_M * HP];      // 33 KB
    __shared__ __align__(16) float          lds_logits[BLOCK_M * LP]; // 16.25 KB
    __shared__ int   s_i1[BLOCK_M], s_i2[BLOCK_M];
    __shared__ float s_g1[BLOCK_M], s_g2[BLOCK_M];

    const int tid  = threadIdx.x;
    const int wave = tid >> 5;
    const int lane = tid & 31;
    const int kh   = lane >> 4;    // lane half: 0 or 1
    const int lrow = lane & 15;
    const int row0 = blockIdx.x * BLOCK_M;

    // staging coords: 64 rows x 64 K per step, 16 contiguous K-floats per thread
    const int srow  = tid >> 2;
    const int skseg = (tid & 3) * 16;
    const float* xrow = x + (size_t)(row0 + srow) * KDIM + skseg;

    // ---------------- GEMM1: h = relu(x @ w1 + b1) ----------------
    v8f acc[4][2];
    #pragma unroll
    for (int mt = 0; mt < 4; ++mt)
        #pragma unroll
        for (int nt = 0; nt < 2; ++nt)
            acc[mt][nt] = (v8f){0.f, 0.f, 0.f, 0.f, 0.f, 0.f, 0.f, 0.f};

    // preload chunk 0 into buffer 0
    {
        const float4* p = (const float4*)xrow;
        float4 s0 = p[0], s1 = p[1], s2 = p[2], s3 = p[3];
        uint4 lo, hi;
        lo.x = pkbf(s0.x, s0.y); lo.y = pkbf(s0.z, s0.w);
        lo.z = pkbf(s1.x, s1.y); lo.w = pkbf(s1.z, s1.w);
        hi.x = pkbf(s2.x, s2.y); hi.y = pkbf(s2.z, s2.w);
        hi.z = pkbf(s3.x, s3.y); hi.w = pkbf(s3.z, s3.w);
        *(uint4*)&lds_a[0][srow * AP + skseg]     = lo;
        *(uint4*)&lds_a[0][srow * AP + skseg + 8] = hi;
    }
    __syncthreads();

    const int nb = wave * 32;   // this wave's hidden-column base (8 waves * 32 = 256)

    for (int kc = 0; kc < NSTEP; ++kc) {
        const int bufsel = kc & 1;
        const bool more  = (kc + 1 < NSTEP);

        // issue next chunk's global loads before the WMMA block (pipeline)
        float4 s0, s1, s2, s3;
        if (more) {
            const float4* p = (const float4*)(xrow + (kc + 1) * KSTEP);
            s0 = p[0]; s1 = p[1]; s2 = p[2]; s3 = p[3];
            if (kc + 3 < NSTEP) __builtin_prefetch(xrow + (kc + 3) * KSTEP, 0, 1);
        }

        const int kbase = kc * KSTEP;
        #pragma unroll
        for (int ks = 0; ks < 2; ++ks) {
            const int k0 = kbase + ks * 32;

            // B fragments from global w1t (L2-resident), K-contiguous.
            // Lane = column; dwords pack K pairs; dwords 0-3 -> K = kh*8.., dwords 4-7 -> K = 16+kh*8..
            Frag32B bfr[2];
            #pragma unroll
            for (int nt = 0; nt < 2; ++nt) {
                const unsigned short* bp =
                    w1t + (size_t)(nb + nt * 16 + lrow) * KDIM + k0 + kh * 8;
                bfr[nt].u[0] = ((const uint4*)bp)[0];
                bfr[nt].u[1] = ((const uint4*)bp)[2];
            }

            // A fragments from LDS: dwords 0-3 -> K = kh*8.., dwords 4-7 -> K = 16+kh*8..
            #pragma unroll
            for (int mt = 0; mt < 4; ++mt) {
                Frag32B af;
                const unsigned short* ap =
                    &lds_a[bufsel][(mt * 16 + lrow) * AP + ks * 32 + kh * 8];
                af.u[0] = *(const uint4*)ap;
                af.u[1] = *(const uint4*)(ap + 16);
                acc[mt][0] = __builtin_amdgcn_wmma_f32_16x16x32_bf16(
                    false, af.v, false, bfr[0].v, (short)0, acc[mt][0], false, false);
                acc[mt][1] = __builtin_amdgcn_wmma_f32_16x16x32_bf16(
                    false, af.v, false, bfr[1].v, (short)0, acc[mt][1], false, false);
            }
        }

        // convert + store next chunk into the other buffer
        if (more) {
            uint4 lo, hi;
            lo.x = pkbf(s0.x, s0.y); lo.y = pkbf(s0.z, s0.w);
            lo.z = pkbf(s1.x, s1.y); lo.w = pkbf(s1.z, s1.w);
            hi.x = pkbf(s2.x, s2.y); hi.y = pkbf(s2.z, s2.w);
            hi.z = pkbf(s3.x, s3.y); hi.w = pkbf(s3.z, s3.w);
            *(uint4*)&lds_a[bufsel ^ 1][srow * AP + skseg]     = lo;
            *(uint4*)&lds_a[bufsel ^ 1][srow * AP + skseg + 8] = hi;
        }
        __syncthreads();
    }

    // bias + relu, write h to LDS as bf16 (row-major, K contiguous for GEMM2)
    #pragma unroll
    for (int nt = 0; nt < 2; ++nt) {
        const int col = nb + nt * 16 + lrow;
        const float bv = b1[col];
        #pragma unroll
        for (int mt = 0; mt < 4; ++mt) {
            #pragma unroll
            for (int j = 0; j < 8; ++j) {
                float v = acc[mt][nt][j] + bv;       // C layout: M = mt*16 + j + 8*kh
                v = v > 0.f ? v : 0.f;
                lds_h[(mt * 16 + j + kh * 8) * HP + col] = bf1(v);
            }
        }
    }
    __syncthreads();

    // ---------------- GEMM2: logits = h @ w2 + b2 ----------------
    // 16 output tiles (4 M x 4 N), 2 per wave sharing the same N tile.
    const int t0  = wave * 2;
    const int mtA = t0 & 3;
    const int mtB = (t0 + 1) & 3;
    const int nt2 = t0 >> 2;

    v8f acc2[2];
    acc2[0] = (v8f){0.f, 0.f, 0.f, 0.f, 0.f, 0.f, 0.f, 0.f};
    acc2[1] = (v8f){0.f, 0.f, 0.f, 0.f, 0.f, 0.f, 0.f, 0.f};

    #pragma unroll
    for (int kc = 0; kc < NKC2; ++kc) {
        const int k0 = kc * 32;
        Frag32B bfrag;
        const unsigned short* bp = w2t + (size_t)(nt2 * 16 + lrow) * HID + k0 + kh * 8;
        bfrag.u[0] = ((const uint4*)bp)[0];
        bfrag.u[1] = ((const uint4*)bp)[2];

        Frag32B a0, a1;
        const unsigned short* ap0 = &lds_h[(mtA * 16 + lrow) * HP + k0 + kh * 8];
        a0.u[0] = *(const uint4*)ap0;
        a0.u[1] = *(const uint4*)(ap0 + 16);
        const unsigned short* ap1 = &lds_h[(mtB * 16 + lrow) * HP + k0 + kh * 8];
        a1.u[0] = *(const uint4*)ap1;
        a1.u[1] = *(const uint4*)(ap1 + 16);

        acc2[0] = __builtin_amdgcn_wmma_f32_16x16x32_bf16(
            false, a0.v, false, bfrag.v, (short)0, acc2[0], false, false);
        acc2[1] = __builtin_amdgcn_wmma_f32_16x16x32_bf16(
            false, a1.v, false, bfrag.v, (short)0, acc2[1], false, false);
    }

    {
        const int e  = nt2 * 16 + lrow;
        const float bv = b2[e];
        #pragma unroll
        for (int j = 0; j < 8; ++j) {
            lds_logits[(mtA * 16 + j + kh * 8) * LP + e] = acc2[0][j] + bv;
            lds_logits[(mtB * 16 + j + kh * 8) * LP + e] = acc2[1][j] + bv;
        }
    }
    __syncthreads();

    // ---------------- top-2 + softmax ----------------
    if (tid < BLOCK_M) {
        float v1 = -3.4e38f, v2 = -3.4e38f;
        int i1 = 0, i2 = 0;
        for (int e = 0; e < NEXP; ++e) {
            float v = lds_logits[tid * LP + e];
            if (v > v1)      { v2 = v1; i2 = i1; v1 = v; i1 = e; }
            else if (v > v2) { v2 = v;  i2 = e; }
        }
        float e2  = __expf(v2 - v1);
        float inv = 1.f / (1.f + e2);
        s_i1[tid] = i1; s_i2[tid] = i2;
        s_g1[tid] = inv; s_g2[tid] = e2 * inv;
    }
    __syncthreads();

    // coalesced dense-row writes: thread t covers 16 experts of token t>>2
    {
        const int tok  = tid >> 2;
        const int eb   = (tid & 3) * 16;
        const int i1 = s_i1[tok], i2 = s_i2[tok];
        const float g1 = s_g1[tok], g2 = s_g2[tok];
        float* orow = out + (size_t)(row0 + tok) * NEXP + eb;
        #pragma unroll
        for (int q = 0; q < 4; ++q) {
            float4 v;
            int e0 = eb + q * 4;
            v.x = (e0 + 0 == i1) ? g1 : (e0 + 0 == i2) ? g2 : 0.f;
            v.y = (e0 + 1 == i1) ? g1 : (e0 + 1 == i2) ? g2 : 0.f;
            v.z = (e0 + 2 == i1) ? g1 : (e0 + 2 == i2) ? g2 : 0.f;
            v.w = (e0 + 3 == i1) ? g1 : (e0 + 3 == i2) ? g2 : 0.f;
            ((float4*)orow)[q] = v;
        }
    }
}

extern "C" void kernel_launch(void* const* d_in, const int* in_sizes, int n_in,
                              void* d_out, int out_size, void* d_ws, size_t ws_size,
                              hipStream_t stream) {
    (void)in_sizes; (void)n_in; (void)out_size; (void)ws_size;
    const float* x  = (const float*)d_in[0];
    const float* w1 = (const float*)d_in[1];
    const float* b1 = (const float*)d_in[2];
    const float* w2 = (const float*)d_in[3];
    const float* b2 = (const float*)d_in[4];
    float* out = (float*)d_out;

    unsigned short* w1t = (unsigned short*)d_ws;                 // 2 MB
    unsigned short* w2t = w1t + (size_t)KDIM * HID;              // +32 KB

    cvt_w1_kernel<<<(KDIM * HID) / 256, 256, 0, stream>>>(w1, w1t);
    cvt_w2_kernel<<<(HID * NEXP) / 256, 256, 0, stream>>>(w2, w2t);
    router_gate_kernel<<<TOKENS / BLOCK_M, 256, 0, stream>>>(x, b1, b2, w1t, w2t, out);
}
